// Sddr_Single_8555574853699
// MI455X (gfx1250) — matrix operand order repaired and checked by reading, not compiled, not run on hardware
//
#include <hip/hip_runtime.h>
#include <hip/hip_bf16.h>

// Problem sizes (from reference)
#define NROWS 65536
#define SDIM  128
#define DIN   512
#define HDIM  1024
#define DOUTD 256

typedef __attribute__((ext_vector_type(16))) _Float16 v16h;
typedef __attribute__((ext_vector_type(8)))  _Float16 v8h;
typedef __attribute__((ext_vector_type(4)))  _Float16 v4h;
typedef __attribute__((ext_vector_type(8)))  float    v8f;
typedef __attribute__((ext_vector_type(4)))  float    v4f;
typedef int i32x4 __attribute__((vector_size(16)));

#define LDA 520   // padded LDS row stride in halves (1040 B == 4 dwords mod 64 banks)

#define AS1 __attribute__((address_space(1)))
#define AS3 __attribute__((address_space(3)))

// ---------------------------------------------------------------------------
// K1a: W1 [DIN][HDIM] f32  ->  W1T [HDIM][DIN] f16
// ---------------------------------------------------------------------------
__global__ __launch_bounds__(256) void cvt_w1_kernel(const float* __restrict__ W1,
                                                     _Float16* __restrict__ W1T) {
    int idx = blockIdx.x * 256 + threadIdx.x;   // 512*1024 = 524288 elements
    int k = idx >> 10;                          // row in W1
    int n = idx & 1023;                         // col in W1
    W1T[(size_t)n * DIN + k] = (_Float16)W1[idx];
}

// ---------------------------------------------------------------------------
// K1b: u2[h] = sum_k W2[h][k] * w_deep[k]
// ---------------------------------------------------------------------------
__global__ __launch_bounds__(256) void u2_kernel(const float* __restrict__ W2,
                                                 const float* __restrict__ wdeep,
                                                 float* __restrict__ u2) {
    int h = blockIdx.x * 256 + threadIdx.x;     // HDIM threads total
    float acc = 0.f;
    for (int k = 0; k < DOUTD; ++k) acc += W2[(size_t)h * DOUTD + k] * wdeep[k];
    u2[h] = acc;
}

// ---------------------------------------------------------------------------
// K1c: c = b2 . w_deep
// ---------------------------------------------------------------------------
__global__ __launch_bounds__(256) void c_kernel(const float* __restrict__ b2,
                                                const float* __restrict__ wdeep,
                                                float* __restrict__ cbuf) {
    __shared__ float red[256];
    int tid = threadIdx.x;
    red[tid] = b2[tid] * wdeep[tid];
    __syncthreads();
    for (int s = 128; s > 0; s >>= 1) {
        if (tid < s) red[tid] += red[tid + s];
        __syncthreads();
    }
    if (tid == 0) cbuf[0] = red[0];
}

// ---------------------------------------------------------------------------
// K2: fused  v[i] = sum_h relu(d1[i,:]@W1[:,h] + b1[h]) * u2[h]
// WMMA f16 GEMM, 128-row slab per block, 8 waves x 16-row strips.
// B tiles (64 cols x 512 K) are double-buffered in LDS and staged with
// GLOBAL_LOAD_ASYNC_TO_LDS_B128 (ASYNCcnt), overlapping with WMMA compute.
// ---------------------------------------------------------------------------
__global__ __launch_bounds__(256) void fused_mlp_kernel(const float* __restrict__ d1,
                                                        const _Float16* __restrict__ W1T,
                                                        const float* __restrict__ bias1,
                                                        const float* __restrict__ u2,
                                                        float* __restrict__ v) {
    extern __shared__ char dyn_smem[];
    _Float16* As = (_Float16*)dyn_smem;                        // [128][LDA]  133,120 B
    _Float16* Bs0 = (_Float16*)(dyn_smem + 128 * LDA * 2);     // [64][LDA]    66,560 B
    _Float16* Bs1 = Bs0 + 64 * LDA;                            // [64][LDA]    66,560 B

    const int tid  = threadIdx.x;
    const int lane = tid & 31;
    const int wave = tid >> 5;
    const int l15  = lane & 15;
    const bool hi  = lane >= 16;
    const int row0 = blockIdx.x * 128;

    // ---- stage A slab: 128 x 512 f32 -> f16 into LDS ----
    for (int it = 0; it < 64; ++it) {
        int flat4 = tid + 256 * it;          // 16384 float4 total
        int r  = flat4 >> 7;                 // 128 float4 per row
        int c4 = flat4 & 127;
        v4f x = *(const v4f*)(d1 + (size_t)(row0 + r) * DIN + c4 * 4);
        v4h h4;
        h4[0] = (_Float16)x.x; h4[1] = (_Float16)x.y;
        h4[2] = (_Float16)x.z; h4[3] = (_Float16)x.w;
        *(v4h*)(As + r * LDA + c4 * 4) = h4;
    }

    // ---- async-stage helper: B tile = W1T cols [ncol0, ncol0+64) x K=512 ----
    // 64 rows x 64 chunks of 16 B = 4096 b128 copies, 16 per thread.
    auto stage_async = [&](int ncol0, _Float16* dst) {
#pragma unroll
        for (int it = 0; it < 16; ++it) {
            int flat16 = tid + 256 * it;
            int n  = flat16 >> 6;            // 0..63
            int c8 = flat16 & 63;            // 0..63 (8-half chunks)
            const _Float16* gp = W1T + (size_t)(ncol0 + n) * DIN + c8 * 8;
            _Float16* lp = dst + n * LDA + c8 * 8;
            __builtin_amdgcn_global_load_async_to_lds_b128((AS1 i32x4*)gp, (AS3 i32x4*)lp, 0, 0);
        }
    };

    float vpart[8] = {0.f, 0.f, 0.f, 0.f, 0.f, 0.f, 0.f, 0.f};
    const int mrow = wave * 16 + l15;                 // A-fragment row for this lane
    const _Float16* arow = As + mrow * LDA + (hi ? 16 : 0);

    // prefetch tile 0
    stage_async(0, Bs0);
    asm volatile("s_wait_asynccnt 0x0" ::: "memory");
    __syncthreads();   // also covers the A-slab DS stores

    for (int nt = 0; nt < 16; ++nt) {
        _Float16* Bcur = (nt & 1) ? Bs1 : Bs0;
        _Float16* Bnxt = (nt & 1) ? Bs0 : Bs1;
        if (nt + 1 < 16) stage_async((nt + 1) * 64, Bnxt);   // overlap with compute

        v8f acc[4];
        const v8f zero = {0.f, 0.f, 0.f, 0.f, 0.f, 0.f, 0.f, 0.f};
#pragma unroll
        for (int i = 0; i < 4; ++i) acc[i] = zero;

        for (int kk = 0; kk < 16; ++kk) {
            const int k0 = kk * 32;
            // A fragment: lanes 0-15 hold K k0..k0+15, lanes 16-31 K+16
            v8h a_lo = *(const v8h*)(arow + k0);
            v8h a_hi = *(const v8h*)(arow + k0 + 8);
            v16h a;
#pragma unroll
            for (int e = 0; e < 8; ++e) { a[e] = a_lo[e]; a[e + 8] = a_hi[e]; }
#pragma unroll
            for (int ns = 0; ns < 4; ++ns) {
                // B fragment: lanes 0-15 hold K k0..k0+15 of col n, lanes 16-31 K+16
                const _Float16* brow = Bcur + (ns * 16 + l15) * LDA + k0 + (hi ? 16 : 0);
                v8h b_lo = *(const v8h*)(brow);
                v8h b_hi = *(const v8h*)(brow + 8);
                v16h b;
#pragma unroll
                for (int e = 0; e < 8; ++e) { b[e] = b_lo[e]; b[e + 8] = b_hi[e]; }
                acc[ns] = __builtin_amdgcn_wmma_f32_16x16x32_f16(
                    false, a, false, b, (short)0, acc[ns], false, false);
            }
        }

        // ---- epilogue: bias + relu + dot with u2 ----
#pragma unroll
        for (int ns = 0; ns < 4; ++ns) {
            int ncol = nt * 64 + ns * 16 + l15;
            float bb = bias1[ncol];
            float uu = u2[ncol];
#pragma unroll
            for (int rr = 0; rr < 8; ++rr) {
                float t = acc[ns][rr] + bb;
                t = t > 0.f ? t : 0.f;
                vpart[rr] += t * uu;
            }
        }

        // ensure our prefetch landed; barrier makes it visible block-wide
        asm volatile("s_wait_asynccnt 0x0" ::: "memory");
        __syncthreads();
    }

    // C layout: lanes 0-15 -> rows rr, lanes 16-31 -> rows rr+8 (N = lane&15)
#pragma unroll
    for (int rr = 0; rr < 8; ++rr) {
        int m = hi ? rr + 8 : rr;
        atomicAdd(v + row0 + wave * 16 + m, vpart[rr]);
    }
}

// ---------------------------------------------------------------------------
// K3: Gram G = S^T S  (f32, register-blocked 8x8 per thread)
// ---------------------------------------------------------------------------
__global__ __launch_bounds__(256) void gram_kernel(const float* __restrict__ S,
                                                   float* __restrict__ G) {
    __shared__ float rows[8][128];
    const int tid = threadIdx.x;
    const int tx = tid & 15, ty = tid >> 4;
    float acc[8][8] = {};
    const int i0 = blockIdx.x * 1024;
    for (int chunk = 0; chunk < 128; ++chunk) {
        int base = i0 + chunk * 8;
        {
            int r  = tid >> 5;               // 32 float4 per row
            int c4 = tid & 31;
            *(v4f*)&rows[r][c4 * 4] = *(const v4f*)(S + (size_t)(base + r) * SDIM + c4 * 4);
        }
        __syncthreads();
#pragma unroll
        for (int rr = 0; rr < 8; ++rr) {
            float sa[8], sb[8];
#pragma unroll
            for (int j = 0; j < 8; ++j) { sa[j] = rows[rr][ty * 8 + j]; sb[j] = rows[rr][tx * 8 + j]; }
#pragma unroll
            for (int i = 0; i < 8; ++i)
#pragma unroll
                for (int j = 0; j < 8; ++j) acc[i][j] += sa[i] * sb[j];
        }
        __syncthreads();
    }
#pragma unroll
    for (int i = 0; i < 8; ++i)
#pragma unroll
        for (int j = 0; j < 8; ++j)
            atomicAdd(&G[(ty * 8 + i) * SDIM + tx * 8 + j], acc[i][j]);
}

// ---------------------------------------------------------------------------
// K4: r = S^T (v + c)
// ---------------------------------------------------------------------------
__global__ __launch_bounds__(256) void stv_kernel(const float* __restrict__ S,
                                                  const float* __restrict__ v,
                                                  const float* __restrict__ cbuf,
                                                  float* __restrict__ r) {
    __shared__ float red[256];
    const int tid = threadIdx.x;
    const int a = tid & 127, half = tid >> 7;
    const float c = cbuf[0];
    float acc = 0.f;
    const int i0 = blockIdx.x * 512;
    for (int i = i0 + half; i < i0 + 512; i += 2)
        acc += S[(size_t)i * SDIM + a] * (v[i] + c);
    red[tid] = acc;
    __syncthreads();
    if (half == 0) atomicAdd(&r[a], red[a] + red[a + 128]);
}

// ---------------------------------------------------------------------------
// K5: Cholesky solve G x = r ; y = w_struct - x   (single block, 128 threads)
// ---------------------------------------------------------------------------
__global__ __launch_bounds__(128) void solve_kernel(const float* __restrict__ G,
                                                    const float* __restrict__ r,
                                                    const float* __restrict__ wstruct,
                                                    float* __restrict__ y) {
    extern __shared__ char dyn_smem[];
    float* g = (float*)dyn_smem;         // [128][128]
    float* z = g + 128 * 128;            // [128]
    const int tid = threadIdx.x;
    for (int j = 0; j < 128; ++j) g[j * 128 + tid] = G[j * 128 + tid];
    z[tid] = r[tid];
    __syncthreads();
    // Cholesky (lower)
    for (int k = 0; k < 128; ++k) {
        if (tid == 0) g[k * 128 + k] = sqrtf(g[k * 128 + k]);
        __syncthreads();
        float dk = g[k * 128 + k];
        if (tid > k) g[tid * 128 + k] /= dk;
        __syncthreads();
        if (tid > k) {
            float lik = g[tid * 128 + k];
            for (int j = k + 1; j <= tid; ++j) g[tid * 128 + j] -= lik * g[j * 128 + k];
        }
        __syncthreads();
    }
    // forward: L z = r
    for (int k = 0; k < 128; ++k) {
        if (tid == 0) z[k] /= g[k * 128 + k];
        __syncthreads();
        if (tid > k) z[tid] -= g[tid * 128 + k] * z[k];
        __syncthreads();
    }
    // backward: L^T x = z
    for (int k = 127; k >= 0; --k) {
        if (tid == 0) z[k] /= g[k * 128 + k];
        __syncthreads();
        if (tid < k) z[tid] -= g[k * 128 + tid] * z[k];
        __syncthreads();
    }
    y[tid] = wstruct[tid] - z[tid];
}

// ---------------------------------------------------------------------------
// K6: out[i] = v[i] + c + S[i,:] . y   (one wave per row iteration)
// ---------------------------------------------------------------------------
__global__ __launch_bounds__(256) void out_kernel(const float* __restrict__ S,
                                                  const float* __restrict__ v,
                                                  const float* __restrict__ cbuf,
                                                  const float* __restrict__ y,
                                                  float* __restrict__ out) {
    const int tid = threadIdx.x;
    const int lane = tid & 31, wv = tid >> 5;
    const float c = cbuf[0];
    v4f y4 = *(const v4f*)(y + lane * 4);
    const int gw = blockIdx.x * 8 + wv;  // 2048 waves, 32 rows each
    for (int it = 0; it < 32; ++it) {
        int i = gw * 32 + it;
        v4f s4 = *(const v4f*)(S + (size_t)i * SDIM + lane * 4);
        float p = s4.x * y4.x + s4.y * y4.y + s4.z * y4.z + s4.w * y4.w;
        for (int off = 16; off > 0; off >>= 1) p += __shfl_xor(p, off, 32);
        if (lane == 0) out[i] = v[i] + c + p;
    }
}

// ---------------------------------------------------------------------------
extern "C" void kernel_launch(void* const* d_in, const int* in_sizes, int n_in,
                              void* d_out, int out_size, void* d_ws, size_t ws_size,
                              hipStream_t stream) {
    const float* S       = (const float*)d_in[0];
    const float* d1      = (const float*)d_in[1];
    const float* W1      = (const float*)d_in[2];
    const float* b1      = (const float*)d_in[3];
    const float* W2      = (const float*)d_in[4];
    const float* b2      = (const float*)d_in[5];
    const float* wstruct = (const float*)d_in[6];
    const float* wdeep   = (const float*)d_in[7];
    float* out = (float*)d_out;
    char* ws = (char*)d_ws;

    _Float16* W1T = (_Float16*)(ws);                 // 1,048,576 B
    float* v    = (float*)(ws + 1048576);            //   262,144 B
    float* u2   = (float*)(ws + 1310720);            //     4,096 B
    float* G    = (float*)(ws + 1314816);            //    65,536 B
    float* r    = (float*)(ws + 1380352);            //       512 B
    float* y    = (float*)(ws + 1380864);            //       512 B
    float* cbuf = (float*)(ws + 1381376);            //       256 B

    (void)hipMemsetAsync(v, 0, NROWS * sizeof(float), stream);
    (void)hipMemsetAsync(G, 0, SDIM * SDIM * sizeof(float), stream);
    (void)hipMemsetAsync(r, 0, SDIM * sizeof(float), stream);

    cvt_w1_kernel<<<(DIN * HDIM) / 256, 256, 0, stream>>>(W1, W1T);
    u2_kernel<<<HDIM / 256, 256, 0, stream>>>(W2, wdeep, u2);
    c_kernel<<<1, 256, 0, stream>>>(b2, wdeep, cbuf);
    gram_kernel<<<64, 256, 0, stream>>>(S, G);

    size_t smem_main = (size_t)(128 + 64 + 64) * LDA * sizeof(_Float16);  // 266,240 B
    fused_mlp_kernel<<<NROWS / 128, 256, smem_main, stream>>>(d1, W1T, b1, u2, v);

    stv_kernel<<<128, 256, 0, stream>>>(S, v, cbuf, r);
    solve_kernel<<<1, 128, (128 * 128 + 128) * sizeof(float), stream>>>(G, r, wstruct, y);
    out_kernel<<<256, 256, 0, stream>>>(S, v, cbuf, y, out);
}
